// Transformer_82583631167783
// MI455X (gfx1250) — matrix operand order repaired and checked by reading, not compiled, not run on hardware
//
#include <hip/hip_runtime.h>

// ---------------- model constants ----------------
#define Hc      768
#define NHc     12
#define HDc     64
#define INTERc  3072
#define NLc     6
#define AWc     256
#define SEQc    512
#define Bc      16
#define NTOK    (Bc * SEQc)          // 8192
#define MIDW    (3 * Hc + INTERc)    // 5376
#define COMBW   (Hc + INTERc)        // 3840

typedef __attribute__((ext_vector_type(16))) __bf16 v16bf;
typedef __attribute__((ext_vector_type(8)))  float  v8f;

union Frag {
    v16bf v;
    unsigned u[8];
    unsigned short s[16];
};

__device__ __forceinline__ unsigned short f2bf(float f) {
    union { float f; unsigned u; } x; x.f = f;
    unsigned r = x.u + 0x7FFFu + ((x.u >> 16) & 1u);   // round-to-nearest-even
    return (unsigned short)(r >> 16);
}
__device__ __forceinline__ v8f wmma_bf16(const Frag& a, const Frag& b, v8f c) {
    return __builtin_amdgcn_wmma_f32_16x16x32_bf16(false, a.v, false, b.v,
                                                   (short)0, c, false, false);
}

// ---- gfx1250 async global->LDS copy (ASYNCcnt-tracked DMA, bypasses VGPRs) ----
__device__ __forceinline__ void async_b128(const void* g, void* l) {
    unsigned lds = (unsigned)(uintptr_t)l;                    // LDS offset = low 32 bits
    unsigned long long ga = (unsigned long long)(uintptr_t)g;
    asm volatile("global_load_async_to_lds_b128 %0, %1, off"
                 :: "v"(lds), "v"(ga) : "memory");
}
__device__ __forceinline__ void wait_async0() {
    asm volatile("s_wait_asynccnt 0x0" ::: "memory");
}

// ---------------- embed gather ----------------
__global__ __launch_bounds__(256) void k_embed(const int* __restrict__ tokens,
                                               const float* __restrict__ embed,
                                               float* __restrict__ x) {
    int row = blockIdx.x;
    int tok = tokens[row];
    const float* src = embed + (size_t)tok * Hc;
    float* dst = x + (size_t)row * Hc;
    for (int c = threadIdx.x; c < Hc; c += 256) dst[c] = src[c];
}

// ---------------- rotary tables (per-pair) ----------------
__global__ __launch_bounds__(256) void k_sincos(const float* __restrict__ ages,
                                                float* __restrict__ st,
                                                float* __restrict__ ct) {
    int idx = blockIdx.x * 256 + threadIdx.x;
    if (idx >= NTOK * 32) return;
    int n = idx >> 5, i = idx & 31;
    float inv = powf(10000.f, -(2.f * (float)i) / 31.f);  // linspace(0,2,32)
    float t = ages[n] * inv;
    st[idx] = sinf(t);
    ct[idx] = cosf(t);
}

// ---------------- f32 -> bf16 conversion + transpose: dst[n][k] = src[k][n] ----------------
__global__ __launch_bounds__(256) void k_cvt_t(const float* __restrict__ src,
                                               unsigned short* __restrict__ dst,
                                               int K, int Nn) {
    long long total = (long long)K * Nn;
    long long idx = (long long)blockIdx.x * 256 + threadIdx.x;
    if (idx >= total) return;
    const float* s = src + (long long)blockIdx.y * total;
    unsigned short* d = dst + (long long)blockIdx.y * total;
    int k = (int)(idx / Nn), n = (int)(idx % Nn);
    d[(size_t)n * K + k] = f2bf(s[idx]);
}

// ---------------- layernorm (writes f32 + optional bf16) ----------------
__global__ __launch_bounds__(256) void k_ln(const float* __restrict__ x,
                                            const float* __restrict__ sc,
                                            const float* __restrict__ off,
                                            float* __restrict__ xn,
                                            unsigned short* __restrict__ xnb) {
    __shared__ float red1[8], red2[8], bc[2];
    int row = blockIdx.x, tid = threadIdx.x;
    int lane = tid & 31, wid = tid >> 5;
    const float* xr = x + (size_t)row * Hc;
    float v[3], s1 = 0.f, s2 = 0.f;
#pragma unroll
    for (int k = 0; k < 3; k++) {
        v[k] = xr[tid + 256 * k];
        s1 += v[k];
        s2 += v[k] * v[k];
    }
#pragma unroll
    for (int o = 16; o >= 1; o >>= 1) {
        s1 += __shfl_xor(s1, o, 32);
        s2 += __shfl_xor(s2, o, 32);
    }
    if (lane == 0) { red1[wid] = s1; red2[wid] = s2; }
    __syncthreads();
    if (tid == 0) {
        float t1 = 0.f, t2 = 0.f;
        for (int w = 0; w < 8; w++) { t1 += red1[w]; t2 += red2[w]; }
        float mean = t1 / (float)Hc;
        float var = t2 / (float)Hc - mean * mean;
        bc[0] = mean;
        bc[1] = rsqrtf(var + 1e-5f);
    }
    __syncthreads();
    float mean = bc[0], rstd = bc[1];
#pragma unroll
    for (int k = 0; k < 3; k++) {
        int c = tid + 256 * k;
        float val = (v[k] - mean) * rstd * sc[c] + off[c];
        xn[(size_t)row * Hc + c] = val;
        if (xnb) xnb[(size_t)row * Hc + c] = f2bf(val);
    }
}

// ---------------- WMMA tiled GEMM ----------------
// C[M x Nn] = A(bf16, M x K, row-major) @ BT(bf16, Nn x K, i.e. B transposed) + bias (+res)
// Block tile 128x128, BK=32, 8 waves each computing 32x64 (2x4 WMMA tiles).
// Double-buffered LDS filled by async global->LDS DMA; prefetch two tiles ahead.
template <bool RES>
__global__ __launch_bounds__(256) void k_gemm(const unsigned short* __restrict__ A,
                                              const unsigned short* __restrict__ BT,
                                              const float* __restrict__ bias,
                                              const float* __restrict__ res,
                                              float* __restrict__ C,
                                              int Nn, int K) {
    __shared__ unsigned short As[2][128][32];
    __shared__ unsigned short Bs[2][128][32];
    const int tid = threadIdx.x, lane = tid & 31, wid = tid >> 5;
    const int hi = lane >> 4, ln = lane & 15;
    const int row0 = blockIdx.y * 128, col0 = blockIdx.x * 128;
    const int wr = (wid & 3) * 32, wc = (wid >> 2) * 64;

    const int fr = tid >> 1;         // 0..127 (tile row for A, tile col for BT)
    const int fc = (tid & 1) * 16;   // 0 / 16 along K
    const unsigned short* Ab = A + (size_t)(row0 + fr) * K + fc;
    const unsigned short* Bb = BT + (size_t)(col0 + fr) * K + fc;

    v8f acc[2][4];
#pragma unroll
    for (int i = 0; i < 2; i++)
#pragma unroll
        for (int j = 0; j < 4; j++) acc[i][j] = (v8f){};

    // kt=0 tile via async DMA
    async_b128(Ab,     &As[0][fr][fc]);
    async_b128(Ab + 8, &As[0][fr][fc + 8]);
    async_b128(Bb,     &Bs[0][fr][fc]);
    async_b128(Bb + 8, &Bs[0][fr][fc + 8]);
    wait_async0();
    __syncthreads();

    int buf = 0;
    for (int kt = 0; kt < K; kt += 32, buf ^= 1) {
        const bool more = (kt + 32) < K;
        if (more) {
            // issue next tile's DMA before computing (overlaps with WMMAs)
            async_b128(Ab + kt + 32, &As[buf ^ 1][fr][fc]);
            async_b128(Ab + kt + 40, &As[buf ^ 1][fr][fc + 8]);
            async_b128(Bb + kt + 32, &Bs[buf ^ 1][fr][fc]);
            async_b128(Bb + kt + 40, &Bs[buf ^ 1][fr][fc + 8]);
            if (kt + 64 < K) {   // gfx1250 global_prefetch two tiles ahead
                __builtin_prefetch(Ab + kt + 64, 0, 1);
                __builtin_prefetch(Bb + kt + 64, 0, 1);
            }
        }

        Frag af[2], bfr[4];
#pragma unroll
        for (int i = 0; i < 2; i++) {
            int rrow = wr + 16 * i + ln;
#pragma unroll
            for (int j = 0; j < 8; j++) {
                int koff = (j < 4) ? (8 * hi + 2 * j) : (16 + 8 * hi + 2 * (j - 4));
                af[i].u[j] = *(const unsigned*)&As[buf][rrow][koff];
            }
        }
#pragma unroll
        for (int j = 0; j < 4; j++) {
            int col = wc + 16 * j + ln;
#pragma unroll
            for (int jv = 0; jv < 8; jv++)
                bfr[j].u[jv] = *(const unsigned*)&Bs[buf][col][(jv < 4) ? (8 * hi + 2 * jv) : (16 + 8 * hi + 2 * (jv - 4))];
        }
#pragma unroll
        for (int i = 0; i < 2; i++)
#pragma unroll
            for (int j = 0; j < 4; j++)
                acc[i][j] = wmma_bf16(af[i], bfr[j], acc[i][j]);

        if (more) {
            wait_async0();     // own DMA done -> LDS
            __syncthreads();   // everyone's DMA visible
        }
    }

    // epilogue: pointer-stepped, branch-free
    float bi[4];
#pragma unroll
    for (int j = 0; j < 4; j++) bi[j] = bias[col0 + wc + 16 * j + ln];
#pragma unroll
    for (int i = 0; i < 2; i++) {
        int grow = row0 + wr + 16 * i + 8 * hi;
        float* cp = C + (size_t)grow * Nn + (col0 + wc + ln);
        const float* rp = RES ? (res + (size_t)grow * Nn + (col0 + wc + ln)) : nullptr;
#pragma unroll
        for (int r = 0; r < 8; r++) {
#pragma unroll
            for (int j = 0; j < 4; j++) {
                float v = acc[i][j][r] + bi[j];
                if (RES) v += rp[16 * j];
                cp[16 * j] = v;
            }
            cp += Nn;
            if (RES) rp += Nn;
        }
    }
}

// ---------------- rope + qkv head pack ----------------
// qh/kh: [head][token][d] bf16 ; vhT: [head][d][token] bf16 (so PV B-frags are packed pairs)
__global__ __launch_bounds__(256) void k_rope(const float* __restrict__ mid,
                                              const float* __restrict__ st,
                                              const float* __restrict__ ct,
                                              unsigned short* __restrict__ qh,
                                              unsigned short* __restrict__ kh,
                                              unsigned short* __restrict__ vhT) {
    int n = blockIdx.x, tid = threadIdx.x;
    const float* mr = mid + (size_t)n * MIDW;
    const float* sp = st + (size_t)n * 32;
    const float* cp = ct + (size_t)n * 32;
    for (int p = tid; p < NHc * 32; p += 256) {   // 384 rotary pairs for q and k
        int head = p >> 5, i = p & 31;
        float c = cp[i], s = sp[i];
        size_t dst = ((size_t)head * NTOK + n) * HDc + 2 * i;
        float q0 = mr[2 * p], q1 = mr[2 * p + 1];
        qh[dst]     = f2bf(q0 * c - q1 * s);
        qh[dst + 1] = f2bf(q1 * c + q0 * s);
        float k0 = mr[Hc + 2 * p], k1 = mr[Hc + 2 * p + 1];
        kh[dst]     = f2bf(k0 * c - k1 * s);
        kh[dst + 1] = f2bf(k1 * c + k0 * s);
    }
    for (int e = tid; e < Hc; e += 256) {         // v, transposed per head
        int head = e >> 6, d = e & 63;
        vhT[((size_t)head * HDc + d) * NTOK + n] = f2bf(mr[2 * Hc + e]);
    }
}

// ---------------- flash attention: window-causal, per-wave 16-query tile ----------------
__global__ __launch_bounds__(256) void k_attn(const unsigned short* __restrict__ qh,
                                              const unsigned short* __restrict__ kh,
                                              const unsigned short* __restrict__ vhT,
                                              unsigned short* __restrict__ comb) {
    __shared__ unsigned short Pb[8][16][32];
    const int tid = threadIdx.x, lane = tid & 31, w = tid >> 5;
    const int hi = lane >> 4, ln = lane & 15;
    const int h = blockIdx.z, b = blockIdx.y, qblk = blockIdx.x;
    const int q0 = (qblk * 8 + w) * 16;
    const int tok0 = b * SEQc;
    const size_t hb  = (size_t)h * NTOK * HDc;   // qh / kh base
    const size_t hbT = (size_t)h * HDc * NTOK;   // vhT base

    Frag qa[2];
#pragma unroll
    for (int f = 0; f < 2; f++) {
        int qrow = tok0 + q0 + ln;
#pragma unroll
        for (int j = 0; j < 8; j++) {
            int koff = ((j < 4) ? (8 * hi + 2 * j) : (16 + 8 * hi + 2 * (j - 4))) + 32 * f;
            qa[f].u[j] = *(const unsigned*)&qh[hb + (size_t)qrow * HDc + koff];
        }
    }

    float m[8], l[8];
    v8f O[4];
#pragma unroll
    for (int r = 0; r < 8; r++) { m[r] = -1e30f; l[r] = 0.f; }
#pragma unroll
    for (int t = 0; t < 4; t++) O[t] = (v8f){};

    int lo = q0 - (AWc - 1);
    if (lo < 0) lo = 0;
    lo &= ~31;
    const float scale = 0.125f;  // 1/sqrt(64)

    for (int kb = lo; kb < q0 + 16; kb += 32) {
        v8f S[2];
#pragma unroll
        for (int sub = 0; sub < 2; sub++) {
            S[sub] = (v8f){};
            Frag kf[2];
            int key = tok0 + kb + sub * 16 + ln;
#pragma unroll
            for (int f = 0; f < 2; f++) {
#pragma unroll
                for (int jv = 0; jv < 8; jv++) {
                    int d = 32 * f + ((jv < 4) ? (8 * hi + 2 * jv) : (16 + 8 * hi + 2 * (jv - 4)));
                    kf[f].u[jv] = *(const unsigned*)&kh[hb + (size_t)key * HDc + d];
                }
            }
            S[sub] = wmma_bf16(qa[0], kf[0], S[sub]);
            S[sub] = wmma_bf16(qa[1], kf[1], S[sub]);
        }

        float alpha[8];
#pragma unroll
        for (int r = 0; r < 8; r++) {
            int iq = q0 + r + 8 * hi;
            float mx = -1e30f;
#pragma unroll
            for (int sub = 0; sub < 2; sub++) {
                int j = kb + sub * 16 + ln;
                float s = S[sub][r] * scale;
                bool ok = (j <= iq) && (iq - j < AWc);
                s = ok ? s : -1e30f;
                S[sub][r] = s;
                mx = fmaxf(mx, s);
            }
#pragma unroll
            for (int o = 1; o <= 8; o <<= 1) mx = fmaxf(mx, __shfl_xor(mx, o, 32));
            float mn = fmaxf(m[r], mx);
            alpha[r] = __expf(m[r] - mn);
            m[r] = mn;
        }

#pragma unroll
        for (int r = 0; r < 8; r++) {
            int iq = q0 + r + 8 * hi;
            float rs = 0.f;
#pragma unroll
            for (int sub = 0; sub < 2; sub++) {
                int j = kb + sub * 16 + ln;
                bool ok = (j <= iq) && (iq - j < AWc);
                float p = ok ? __expf(S[sub][r] - m[r]) : 0.f;
                rs += p;
                Pb[w][r + 8 * hi][sub * 16 + ln] = f2bf(p);
            }
#pragma unroll
            for (int o = 1; o <= 8; o <<= 1) rs += __shfl_xor(rs, o, 32);
            l[r] = l[r] * alpha[r] + rs;
        }
#pragma unroll
        for (int t = 0; t < 4; t++)
#pragma unroll
            for (int r = 0; r < 8; r++) O[t][r] *= alpha[r];

        // P (16x32) as A fragment from LDS
        Frag pa;
#pragma unroll
        for (int j = 0; j < 8; j++) {
            int koff = (j < 4) ? (8 * hi + 2 * j) : (16 + 8 * hi + 2 * (j - 4));
            pa.u[j] = *(const unsigned*)&Pb[w][ln][koff];
        }
        // O += P @ V : B-frag pairs (kk,kk+1) contiguous in vhT
#pragma unroll
        for (int t = 0; t < 4; t++) {
            Frag vb;
            const unsigned short* vrow = vhT + hbT + (size_t)(t * 16 + ln) * NTOK;
#pragma unroll
            for (int jv = 0; jv < 8; jv++) {
                int kk = tok0 + kb + ((jv < 4) ? (8 * hi + 2 * jv) : (16 + 8 * hi + 2 * (jv - 4)));
                vb.u[jv] = *(const unsigned*)&vrow[kk];
            }
            O[t] = wmma_bf16(pa, vb, O[t]);
        }
    }

#pragma unroll
    for (int t = 0; t < 4; t++)
#pragma unroll
        for (int r = 0; r < 8; r++) {
            int row = tok0 + q0 + r + 8 * hi;
            int col = h * HDc + t * 16 + ln;
            comb[(size_t)row * COMBW + col] = f2bf(O[t][r] / l[r]);
        }
}

// ---------------- gelu(ff) -> comb bf16 ----------------
__global__ __launch_bounds__(256) void k_gelu(const float* __restrict__ mid,
                                              unsigned short* __restrict__ comb) {
    long long idx = (long long)blockIdx.x * 256 + threadIdx.x;
    if (idx >= (long long)NTOK * INTERc) return;
    int n = (int)(idx / INTERc), c = (int)(idx % INTERc);
    float x = mid[(size_t)n * MIDW + 3 * Hc + c];
    float g = 0.5f * x * (1.f + tanhf(0.7978845608f * (x + 0.044715f * x * x * x)));
    comb[(size_t)n * COMBW + Hc + c] = f2bf(g);
}

// ---------------- host-side orchestration ----------------
extern "C" void kernel_launch(void* const* d_in, const int* in_sizes, int n_in,
                              void* d_out, int out_size, void* d_ws, size_t ws_size,
                              hipStream_t stream) {
    (void)in_sizes; (void)n_in; (void)out_size; (void)ws_size;
    const int*   tokens    = (const int*)d_in[0];
    const float* ages      = (const float*)d_in[1];
    const float* embed     = (const float*)d_in[4];
    const float* ln_scale  = (const float*)d_in[5];
    const float* ln_offset = (const float*)d_in[6];
    const float* w_in      = (const float*)d_in[7];
    const float* b_in      = (const float*)d_in[8];
    const float* w_out     = (const float*)d_in[9];
    const float* b_out     = (const float*)d_in[10];
    const float* fin_scale = (const float*)d_in[11];
    const float* fin_off   = (const float*)d_in[12];
    float* out = (float*)d_out;

    size_t off = 0;
    auto carve = [&](size_t bytes) -> char* {
        char* p = (char*)d_ws + off;
        off += (bytes + 255) & ~(size_t)255;
        return p;
    };
    float*          x       = (float*)carve((size_t)NTOK * Hc * 4);
    float*          xn      = (float*)carve((size_t)NTOK * Hc * 4);
    unsigned short* xnb     = (unsigned short*)carve((size_t)NTOK * Hc * 2);
    float*          mid     = (float*)carve((size_t)NTOK * MIDW * 4);
    unsigned short* comb    = (unsigned short*)carve((size_t)NTOK * COMBW * 2);
    unsigned short* qh      = (unsigned short*)carve((size_t)NHc * NTOK * HDc * 2);
    unsigned short* kh      = (unsigned short*)carve((size_t)NHc * NTOK * HDc * 2);
    unsigned short* vhT     = (unsigned short*)carve((size_t)NHc * HDc * NTOK * 2);
    float*          st      = (float*)carve((size_t)NTOK * 32 * 4);
    float*          ct      = (float*)carve((size_t)NTOK * 32 * 4);
    unsigned short* w_in_bT = (unsigned short*)carve((size_t)NLc * Hc * MIDW * 2);
    unsigned short* w_out_bT= (unsigned short*)carve((size_t)NLc * COMBW * Hc * 2);

    k_embed<<<NTOK, 256, 0, stream>>>(tokens, embed, x);
    k_sincos<<<(NTOK * 32 + 255) / 256, 256, 0, stream>>>(ages, st, ct);
    {
        long long n1 = (long long)Hc * MIDW;        // per-layer elements
        long long n2 = (long long)COMBW * Hc;
        k_cvt_t<<<dim3((unsigned)((n1 + 255) / 256), NLc), 256, 0, stream>>>(
            w_in, w_in_bT, Hc, MIDW);
        k_cvt_t<<<dim3((unsigned)((n2 + 255) / 256), NLc), 256, 0, stream>>>(
            w_out, w_out_bT, COMBW, Hc);
    }

    for (int L = 0; L < NLc; L++) {
        k_ln<<<NTOK, 256, 0, stream>>>(x, ln_scale + (size_t)L * Hc,
                                       ln_offset + (size_t)L * Hc, xn, xnb);
        // mid = xn @ w_in + b_in
        k_gemm<false><<<dim3(MIDW / 128, NTOK / 128), 256, 0, stream>>>(
            xnb, w_in_bT + (size_t)L * Hc * MIDW, b_in + (size_t)L * MIDW,
            nullptr, mid, MIDW, Hc);
        k_rope<<<NTOK, 256, 0, stream>>>(mid, st, ct, qh, kh, vhT);
        k_attn<<<dim3(SEQc / 128, Bc, NHc), 256, 0, stream>>>(qh, kh, vhT, comb);
        k_gelu<<<(unsigned)(((long long)NTOK * INTERc + 255) / 256), 256, 0, stream>>>(mid, comb);
        // x = xn + comb @ w_out + b_out
        k_gemm<true><<<dim3(Hc / 128, NTOK / 128), 256, 0, stream>>>(
            comb, w_out_bT + (size_t)L * COMBW * Hc, b_out + (size_t)L * Hc,
            xn, x, Hc, COMBW);
    }
    k_ln<<<NTOK, 256, 0, stream>>>(x, fin_scale, fin_off, out, nullptr);
}
